// DeepMuGIF_45681272160410
// MI455X (gfx1250) — compile-verified, matrix operand order checked by reading, not run on hardware
//
#include <hip/hip_runtime.h>

// DeepMuGIF forward on MI455X (gfx1250, wave32).
// Heavy convs (1x1 / 3x3 channel GEMMs, ~30 GFLOP) -> V_WMMA_F32_16X16X32_F16.
// Bandwidth-bound ops (depthwise, reductions, softmax, attn*V, FAC) -> VALU.
//
// GEMM structure:
//  * weights pre-packed to zero-padded f16 -> A fragments are 2x
//    global_load_b128 per lane, no guards, no LDS on the A path
//  * activations staged transposed in LDS (Bt[n][k], 80B pitch): each thread
//    owns column n = tid, walks K in batches of 8 (loads clause-grouped into
//    registers, then one ds_store_b128) -> no exec-guard waterfalls, no
//    per-load waits
//  * B fragments gathered as 2x ds_load_b128
//  * tile M=16 x N=128 per 128-thread block; 2 WMMA per K chunk per wave
//  * 3x3 conv = 9 taps x (32+16) K chunks; weights reordered [co][tap][ci|64]

typedef __attribute__((ext_vector_type(16))) _Float16 v16h;
typedef __attribute__((ext_vector_type(8)))  _Float16 v8h;
typedef __attribute__((ext_vector_type(8)))  float    v8f;

#define HEADS 6
#define KW 7
#define CIN 48
#define LP 40       // LDS row pitch in halfs (80 B, multiple of 16 B)

__device__ __forceinline__ v16h frag_pair(v8h lo, v8h up) {
  return __builtin_shufflevector(lo, up, 0, 1, 2, 3, 4, 5, 6, 7,
                                 8, 9, 10, 11, 12, 13, 14, 15);
}

// Fragment element j -> K = 16*(j>>3) + 8*hi + (j&7), hi = lane>>4.
__device__ __forceinline__ v16h frag_from_row(const _Float16* row, int hi) {
  const v8h* r = (const v8h*)row;
  return frag_pair(r[hi], r[2 + hi]);
}

// Stage KV K-rows of one 128-px line into Bt (transposed). Thread owns
// column n; src = element (k=0, column n); K stride = strideK floats.
template <int KV>
__device__ __forceinline__ void stage_cols(_Float16 (*Bt)[LP],
                                           const float* __restrict__ src,
                                           size_t strideK, int n, bool ok) {
#pragma unroll
  for (int kk = 0; kk < KV; kk += 8) {
    float vv[8];
#pragma unroll
    for (int u = 0; u < 8; ++u) vv[u] = src[(size_t)(kk + u) * strideK];
    v8h hv;
#pragma unroll
    for (int u = 0; u < 8; ++u) hv[u] = (_Float16)(ok ? vv[u] : 0.f);
    *(v8h*)&Bt[n][kk] = hv;                       // one ds_store_b128
  }
}

__device__ __forceinline__ void zero_tail16(_Float16 (*Bt)[LP], int n) {
  v8h z = {};
  *(v8h*)&Bt[n][16] = z;
  *(v8h*)&Bt[n][24] = z;
}

// ---------------------------------------------------------------------------
// Weight packing: f32 [Cout][Cin] -> f16 [Mpad][Kpad], zero padded.
// ---------------------------------------------------------------------------
__global__ void pack_w_pw(const float* __restrict__ W, _Float16* __restrict__ Wp,
                          int Cout, int Cin, int Mpad, int Kpad) {
  int i = blockIdx.x * blockDim.x + threadIdx.x;
  if (i >= Mpad * Kpad) return;
  int m = i / Kpad, k = i - m * Kpad;
  float v = (m < Cout && k < Cin) ? W[(size_t)m * Cin + k] : 0.f;
  Wp[i] = (_Float16)v;
}

// f32 OIHW [Cout][48][3][3] -> f16 [Mpad][9][64]: Wp[m][t*64+ci], zero padded.
__global__ void pack_w_conv3(const float* __restrict__ W, _Float16* __restrict__ Wp,
                             int Cout, int Mpad) {
  int i = blockIdx.x * blockDim.x + threadIdx.x;
  if (i >= Mpad * 576) return;
  int m = i / 576, r = i - m * 576;
  int t = r >> 6, ci = r & 63;
  float v = (m < Cout && ci < CIN) ? W[((size_t)m * CIN + ci) * 9 + t] : 0.f;
  Wp[i] = (_Float16)v;
}

// ---------------------------------------------------------------------------
// Pointwise (1x1) conv GEMM: Y[b,co,n] = sum_ci Wp[co,ci]*X[b,ci,n] + bias.
// Wp: packed f16 [Mpad][64]. Block 128 thr; tile M=16 x N=128; b = grid.z.
// ---------------------------------------------------------------------------
template <int COUT, int DO_LRELU>
__global__ __launch_bounds__(128)
void pw_gemm(const float* __restrict__ X, const _Float16* __restrict__ Wp,
             const float* __restrict__ bias, float* __restrict__ Y, int HW) {
  constexpr int KPAD = 64;
  constexpr int MPAD = (COUT + 15) & ~15;
  __shared__ __align__(16) _Float16 Bt[128][LP];   // [n][k], transposed
  const int tid  = threadIdx.x;
  const int lane = tid & 31;
  const int wave = tid >> 5;
  const int hi   = lane >> 4;
  const int mi   = lane & 15;
  const int n0   = blockIdx.x * 128;
  const int m0   = blockIdx.y * 16;
  const int b    = blockIdx.z;
  const float* Xb = X + (size_t)b * CIN * HW + n0 + tid;   // thread's column
  const _Float16* wrow = Wp + (size_t)(m0 + mi) * KPAD;

  v8f acc0 = {}, acc1 = {};
#pragma unroll
  for (int kc = 0; kc < 2; ++kc) {
    const int k0 = kc << 5;
    if (kc == 0) {
      stage_cols<32>(Bt, Xb, HW, tid, true);
    } else {
      stage_cols<16>(Bt, Xb + (size_t)32 * HW, HW, tid, true);
      zero_tail16(Bt, tid);
    }
    __syncthreads();
    v16h a  = frag_from_row(wrow + k0, hi);          // 2x global_load_b128
    const int nl0 = (wave << 5) + mi;
    v16h b0 = frag_from_row(&Bt[nl0][0], hi);        // 2x ds_load_b128
    v16h b1 = frag_from_row(&Bt[nl0 + 16][0], hi);
    acc0 = __builtin_amdgcn_wmma_f32_16x16x32_f16(false, a, false, b0,
                                                  (short)0, acc0, false, false);
    acc1 = __builtin_amdgcn_wmma_f32_16x16x32_f16(false, a, false, b1,
                                                  (short)0, acc1, false, false);
    __syncthreads();
  }
  const int ncol0 = n0 + (wave << 5) + mi;
#pragma unroll
  for (int r = 0; r < 8; ++r) {
    int row = m0 + r + (hi << 3);
    if (MPAD == COUT || row < COUT) {
      float bv = bias[row];
      float v0 = acc0[r] + bv;
      float v1 = acc1[r] + bv;
      if (DO_LRELU) {
        v0 = (v0 > 0.f) ? v0 : 0.1f * v0;
        v1 = (v1 > 0.f) ? v1 : 0.1f * v1;
      }
      float* yr = Y + ((size_t)b * COUT + row) * HW;
      yr[ncol0]      = v0;
      yr[ncol0 + 16] = v1;
    }
  }
}

// ---------------------------------------------------------------------------
// 3x3 conv (zero-pad SAME) GEMM: 9 taps x (32+16) K chunks of Cin.
// Wp: packed f16 [Mpad][9][64]. N tile (128 px) lies inside one image row.
// ---------------------------------------------------------------------------
template <int COUT, int DO_LRELU>
__global__ __launch_bounds__(128)
void conv3_gemm(const float* __restrict__ X, const _Float16* __restrict__ Wp,
                const float* __restrict__ bias, float* __restrict__ Y,
                int Hh, int Ww) {
  constexpr int MPAD = (COUT + 15) & ~15;
  __shared__ __align__(16) _Float16 Bt[128][LP];
  const int HW   = Hh * Ww;
  const int tid  = threadIdx.x;
  const int lane = tid & 31;
  const int wave = tid >> 5;
  const int hi   = lane >> 4;
  const int mi   = lane & 15;
  const int n0   = blockIdx.x * 128;
  const int m0   = blockIdx.y * 16;
  const int b    = blockIdx.z;
  const float* Xb = X + (size_t)b * CIN * HW;
  const _Float16* wrow = Wp + (size_t)(m0 + mi) * 576;
  const int y  = n0 / Ww;                 // tile-constant image row
  const int x0 = n0 - y * Ww;

  v8f acc0 = {}, acc1 = {};
  for (int t = 0; t < 9; ++t) {
    const int dy = t / 3 - 1, dx = t - (t / 3) * 3 - 1;
    const int yy = y + dy;
    const bool yok = (yy >= 0) && (yy < Hh);
    const int yc = min(max(yy, 0), Hh - 1);
    // Thread's shifted pixel (clamped) within row yc.
    const int xx = x0 + tid + dx;
    const int xc = min(max(xx, 0), Ww - 1);
    const bool ok = yok && (xx >= 0) && (xx < Ww);
    const float* src = Xb + (size_t)yc * Ww + xc;   // + ci*HW per K row
#pragma unroll
    for (int sub = 0; sub < 2; ++sub) {
      if (sub == 0) {
        stage_cols<32>(Bt, src, HW, tid, ok);
      } else {
        stage_cols<16>(Bt, src + (size_t)32 * HW, HW, tid, ok);
        zero_tail16(Bt, tid);
      }
      __syncthreads();
      v16h a  = frag_from_row(wrow + t * 64 + (sub << 5), hi);
      const int nl0 = (wave << 5) + mi;
      v16h b0 = frag_from_row(&Bt[nl0][0], hi);
      v16h b1 = frag_from_row(&Bt[nl0 + 16][0], hi);
      acc0 = __builtin_amdgcn_wmma_f32_16x16x32_f16(false, a, false, b0,
                                                    (short)0, acc0, false, false);
      acc1 = __builtin_amdgcn_wmma_f32_16x16x32_f16(false, a, false, b1,
                                                    (short)0, acc1, false, false);
      __syncthreads();
    }
  }
  const int ncol0 = n0 + (wave << 5) + mi;
#pragma unroll
  for (int r = 0; r < 8; ++r) {
    int row = m0 + r + (hi << 3);
    if (MPAD == COUT || row < COUT) {
      float bv = bias[row];
      float v0 = acc0[r] + bv;
      float v1 = acc1[r] + bv;
      if (DO_LRELU) {
        v0 = (v0 > 0.f) ? v0 : 0.1f * v0;
        v1 = (v1 > 0.f) ? v1 : 0.1f * v1;
      }
      float* yr = Y + ((size_t)b * COUT + row) * HW;
      yr[ncol0]      = v0;
      yr[ncol0 + 16] = v1;
    }
  }
}

// ---------------------------------------------------------------------------
// Depthwise 3x3 (zero-pad SAME), per-channel weight + bias.
// ---------------------------------------------------------------------------
__global__ void depthwise3(const float* __restrict__ X, const float* __restrict__ Wd,
                           const float* __restrict__ bias, float* __restrict__ Y,
                           int C, int Hh, int Ww, int total) {
  int idx = blockIdx.x * blockDim.x + threadIdx.x;
  if (idx >= total) return;
  int HW = Hh * Ww;
  int p  = idx % HW;
  int c  = (idx / HW) % C;
  int b  = idx / (HW * C);
  int y = p / Ww, x = p - y * Ww;
  const float* Xc = X + ((size_t)b * C + c) * HW;
  const float* wk = Wd + c * 9;
  float s = bias[c];
#pragma unroll
  for (int dy = -1; dy <= 1; ++dy)
#pragma unroll
    for (int dx = -1; dx <= 1; ++dx) {
      int yy = y + dy, xx = x + dx;
      int yc = min(max(yy, 0), Hh - 1);
      int xc = min(max(xx, 0), Ww - 1);
      float v = Xc[yc * Ww + xc];
      bool ok = (yy >= 0) & (yy < Hh) & (xx >= 0) & (xx < Ww);
      s += (ok ? v : 0.f) * wk[(dy + 1) * 3 + (dx + 1)];
    }
  Y[idx] = s;
}

__global__ void zero_f32(float* p, int n) {
  int i = blockIdx.x * blockDim.x + threadIdx.x;
  if (i < n) p[i] = 0.f;
}

// ---------------------------------------------------------------------------
// Row sum-of-squares over HW (L2 norm). row = b*C + c; channel stride CS.
// ---------------------------------------------------------------------------
__global__ void rowsumsq(const float* __restrict__ X, float* __restrict__ out,
                         int C, int CS, int HW) {
  int row = blockIdx.x;
  int b = row / C, c = row % C;
  const float* p = X + ((size_t)b * CS + c) * HW;
  int per = HW / gridDim.y;
  int beg = blockIdx.y * per, end = beg + per;
  float s = 0.f;
  for (int i = beg + threadIdx.x; i < end; i += blockDim.x) s += p[i] * p[i];
  __shared__ float red[256];
  red[threadIdx.x] = s;
  __syncthreads();
  for (int o = 128; o; o >>= 1) {
    if (threadIdx.x < o) red[threadIdx.x] += red[threadIdx.x + o];
    __syncthreads();
  }
  if (threadIdx.x == 0) atomicAdd(out + row, red[0]);
}

// ---------------------------------------------------------------------------
// qk[b,h,c,d] = sum_n q[b,h*8+c,n] * k[b,h*8+d,n]; q stride QS, k stride KS.
// ---------------------------------------------------------------------------
__global__ void qk_reduce(const float* __restrict__ Q, const float* __restrict__ K,
                          float* __restrict__ qk, int QS, int KS, int HW) {
  int o = blockIdx.x;           // ((b*6 + h)*8 + c)*8 + d
  int d = o & 7;
  int c = (o >> 3) & 7;
  int h = (o >> 6) % HEADS;
  int b = o / (HEADS * 64);
  const float* qp = Q + ((size_t)b * QS + h * 8 + c) * HW;
  const float* kp = K + ((size_t)b * KS + h * 8 + d) * HW;
  int per = HW / gridDim.y;
  int beg = blockIdx.y * per, end = beg + per;
  float s = 0.f;
  for (int i = beg + threadIdx.x; i < end; i += blockDim.x) s += qp[i] * kp[i];
  __shared__ float red[256];
  red[threadIdx.x] = s;
  __syncthreads();
  for (int off = 128; off; off >>= 1) {
    if (threadIdx.x < off) red[threadIdx.x] += red[threadIdx.x + off];
    __syncthreads();
  }
  if (threadIdx.x == 0) atomicAdd(qk + o, red[0]);
}

// ---------------------------------------------------------------------------
// Per-(b,h,c): normalize by L2 norms, scale by temperature, softmax over d.
// ---------------------------------------------------------------------------
__global__ void attn_softmax(const float* __restrict__ qk,
                             const float* __restrict__ ssq,
                             const float* __restrict__ ssk,
                             const float* __restrict__ temp,
                             float* __restrict__ attn, int B) {
  int t = blockIdx.x * blockDim.x + threadIdx.x;
  int total = B * HEADS * 8;
  if (t >= total) return;
  int c = t & 7;
  int h = (t >> 3) % HEADS;
  int b = t / (HEADS * 8);
  float qn = fmaxf(sqrtf(ssq[b * 48 + h * 8 + c]), 1e-12f);
  float tp = temp[h];
  float l[8];
  float mx = -3.4e38f;
#pragma unroll
  for (int d = 0; d < 8; ++d) {
    float kn = fmaxf(sqrtf(ssk[b * 48 + h * 8 + d]), 1e-12f);
    l[d] = qk[(size_t)t * 8 + d] / (qn * kn) * tp;
    mx = fmaxf(mx, l[d]);
  }
  float sum = 0.f;
#pragma unroll
  for (int d = 0; d < 8; ++d) { l[d] = __expf(l[d] - mx); sum += l[d]; }
  float inv = 1.f / sum;
#pragma unroll
  for (int d = 0; d < 8; ++d) attn[(size_t)t * 8 + d] = l[d] * inv;
}

// ---------------------------------------------------------------------------
// out[b,h*8+c,n] = sum_d attn[b,h,c,d] * v[b, VOFF+h*8+d, n] (v stride VS)
// ---------------------------------------------------------------------------
__global__ void attn_apply(const float* __restrict__ attn, const float* __restrict__ V,
                           float* __restrict__ Y, int VS, int VOFF, int HW, int total) {
  int idx = blockIdx.x * blockDim.x + threadIdx.x;
  if (idx >= total) return;
  int n = idx % HW;
  int c = (idx / HW) % 48;
  int b = idx / (HW * 48);
  int h = c >> 3, ci = c & 7;
  const float* ar = attn + (((size_t)b * HEADS + h) * 8 + ci) * 8;
  const float* vb = V + ((size_t)b * VS + VOFF + h * 8) * HW + n;
  float s = 0.f;
#pragma unroll
  for (int d = 0; d < 8; ++d) s += ar[d] * vb[(size_t)d * HW];
  Y[idx] = s;
}

// ---------------------------------------------------------------------------
// FAC: per-pixel dynamic 7x7 conv, edge-clamped padding, kernel shared over c.
// ---------------------------------------------------------------------------
__global__ void fac7(const float* __restrict__ F, const float* __restrict__ Ker,
                     float* __restrict__ Y, int C, int Hh, int Ww, int total) {
  int idx = blockIdx.x * blockDim.x + threadIdx.x;
  if (idx >= total) return;
  int HW = Hh * Ww;
  int p = idx % HW;
  int c = (idx / HW) % C;
  int b = idx / (HW * C);
  int y = p / Ww, x = p - y * Ww;
  const float* Fc = F + ((size_t)b * C + c) * HW;
  const float* Kb = Ker + (size_t)b * (KW * KW) * HW + p;
  float s = 0.f;
#pragma unroll
  for (int i = 0; i < KW; ++i) {
    int yy = min(max(y + i - 3, 0), Hh - 1);
#pragma unroll
    for (int j = 0; j < KW; ++j) {
      int xx = min(max(x + j - 3, 0), Ww - 1);
      s += Fc[yy * Ww + xx] * Kb[(size_t)(i * KW + j) * HW];
    }
  }
  Y[idx] = s;
}

// ---------------------------------------------------------------------------
// Host orchestration
// ---------------------------------------------------------------------------
extern "C" void kernel_launch(void* const* d_in, const int* in_sizes, int n_in,
                              void* d_out, int out_size, void* d_ws, size_t ws_size,
                              hipStream_t stream) {
  (void)in_sizes; (void)n_in; (void)out_size; (void)ws_size;
  const int B = 2, C = 48, H = 256, W = 256, HW = H * W;

  const float* feat[2] = { (const float*)d_in[0], (const float*)d_in[1] };
  const float* temp    = (const float*)d_in[2];
  const float* q_w = (const float*)d_in[3],  * q_b = (const float*)d_in[4];
  const float* qd_w = (const float*)d_in[5], * qd_b = (const float*)d_in[6];
  const float* kv_w = (const float*)d_in[7], * kv_b = (const float*)d_in[8];
  const float* kvd_w = (const float*)d_in[9], * kvd_b = (const float*)d_in[10];
  const float* pr_w = (const float*)d_in[11], * pr_b = (const float*)d_in[12];
  const float* kpw1[2] = { (const float*)d_in[13], (const float*)d_in[19] };
  const float* kpb1[2] = { (const float*)d_in[14], (const float*)d_in[20] };
  const float* kpw2[2] = { (const float*)d_in[15], (const float*)d_in[21] };
  const float* kpb2[2] = { (const float*)d_in[16], (const float*)d_in[22] };
  const float* kpw3[2] = { (const float*)d_in[17], (const float*)d_in[23] };
  const float* kpb3[2] = { (const float*)d_in[18], (const float*)d_in[24] };
  const float* outw[2] = { (const float*)d_in[25], (const float*)d_in[27] };
  const float* outb[2] = { (const float*)d_in[26], (const float*)d_in[28] };
  float* out = (float*)d_out;

  // Workspace carve-up, 256B-aligned blocks.
  char* ws = (char*)d_ws;
  size_t cur = 0;
  auto carveB = [&](size_t bytes) {
    void* p = ws + cur;
    cur = (cur + bytes + 255) & ~(size_t)255;
    return p;
  };
  auto carveF = [&](size_t n) { return (float*)carveB(n * sizeof(float)); };
  auto carveH = [&](size_t n) { return (_Float16*)carveB(n * sizeof(_Float16)); };

  const size_t plane = (size_t)B * C * HW;
  float* bufA  = carveF((size_t)B * 2 * C * HW);   // kv raw (96 ch)
  float* bufB  = carveF((size_t)B * 2 * C * HW);   // kv after depthwise
  float* buf0  = carveF(plane);
  float* buf1  = carveF(plane);
  float* buf2  = carveF(plane);
  float* kbuf  = carveF((size_t)B * KW * KW * HW); // 49-channel kernels
  float* stats = carveF(2 * B * C + 2 * B * HEADS * 64);
  float* ssq  = stats;
  float* ssk  = stats + B * C;
  float* qk   = ssk + B * C;
  float* attn = qk + B * HEADS * 64;
  const int nzero = 2 * B * C + B * HEADS * 64;    // ssq+ssk+qk contiguous

  // Packed f16 weights.
  _Float16* qW16   = carveH(48 * 64);
  _Float16* kvW16  = carveH(96 * 64);
  _Float16* prW16  = carveH(48 * 64);
  _Float16* kp3W16[2] = { carveH(64 * 64), carveH(64 * 64) };
  _Float16* outW16[2] = { carveH(48 * 64), carveH(48 * 64) };
  _Float16* kp1W16[2] = { carveH(48 * 576), carveH(48 * 576) };
  _Float16* kp2W16[2] = { carveH(48 * 576), carveH(48 * 576) };

  const int TPB = 256;
  auto gb = [&](int n) { return (n + TPB - 1) / TPB; };
  pack_w_pw<<<gb(48 * 64), TPB, 0, stream>>>(q_w,  qW16,  48, 48, 48, 64);
  pack_w_pw<<<gb(96 * 64), TPB, 0, stream>>>(kv_w, kvW16, 96, 48, 96, 64);
  pack_w_pw<<<gb(48 * 64), TPB, 0, stream>>>(pr_w, prW16, 48, 48, 48, 64);
  for (int s = 0; s < 2; ++s) {
    pack_w_pw<<<gb(64 * 64), TPB, 0, stream>>>(kpw3[s], kp3W16[s], 49, 48, 64, 64);
    pack_w_pw<<<gb(48 * 64), TPB, 0, stream>>>(outw[s], outW16[s], 48, 48, 48, 64);
    pack_w_conv3<<<gb(48 * 576), TPB, 0, stream>>>(kpw1[s], kp1W16[s], 48, 48);
    pack_w_conv3<<<gb(48 * 576), TPB, 0, stream>>>(kpw2[s], kp2W16[s], 48, 48);
  }

  const dim3 g48(HW / 128, 3, B);   // Cout 48
  const dim3 g96(HW / 128, 6, B);   // Cout 96
  const dim3 g49(HW / 128, 4, B);   // Cout 49 (Mpad 64)
  const int nPlane = (int)plane;
  const int planeBlocks = gb(nPlane);

  for (int s = 0; s < 2; ++s) {
    const float* fg = feat[s];
    const float* fo = feat[1 - s];
    float* ostream = out + (size_t)s * plane;

    // ---- TCA(fg, fo)
    pw_gemm<48, 0><<<g48, 128, 0, stream>>>(fg, qW16, q_b, buf0, HW);
    depthwise3<<<planeBlocks, TPB, 0, stream>>>(buf0, qd_w, qd_b, buf1, C, H, W, nPlane);
    pw_gemm<96, 0><<<g96, 128, 0, stream>>>(fo, kvW16, kv_b, bufA, HW);
    {
      int tot = (int)((size_t)B * 2 * C * HW);
      depthwise3<<<gb(tot), TPB, 0, stream>>>(bufA, kvd_w, kvd_b, bufB, 2 * C, H, W, tot);
    }
    zero_f32<<<gb(nzero), TPB, 0, stream>>>(stats, nzero);
    rowsumsq<<<dim3(B * C, 16), TPB, 0, stream>>>(buf1, ssq, C, C, HW);
    rowsumsq<<<dim3(B * C, 16), TPB, 0, stream>>>(bufB, ssk, C, 2 * C, HW);
    qk_reduce<<<dim3(B * HEADS * 64, 16), TPB, 0, stream>>>(buf1, bufB, qk, C, 2 * C, HW);
    attn_softmax<<<1, 128, 0, stream>>>(qk, ssq, ssk, temp, attn, B);
    attn_apply<<<planeBlocks, TPB, 0, stream>>>(attn, bufB, buf0, 2 * C, C, HW, nPlane);
    pw_gemm<48, 0><<<g48, 128, 0, stream>>>(buf0, prW16, pr_b, buf2, HW);

    // ---- kernel prediction: lrelu(conv3) -> lrelu(conv3) -> conv1x1 C->49
    conv3_gemm<48, 1><<<g48, 128, 0, stream>>>(buf2, kp1W16[s], kpb1[s], buf0, H, W);
    conv3_gemm<48, 1><<<g48, 128, 0, stream>>>(buf0, kp2W16[s], kpb2[s], buf2, H, W);
    pw_gemm<49, 0><<<g49, 128, 0, stream>>>(buf2, kp3W16[s], kpb3[s], kbuf, HW);

    // ---- FAC + output conv (+lrelu)
    fac7<<<planeBlocks, TPB, 0, stream>>>(fg, kbuf, buf0, C, H, W, nPlane);
    pw_gemm<48, 1><<<g48, 128, 0, stream>>>(buf0, outW16[s], outb[s], ostream, HW);
  }
}
